// Experts_33045478375624
// MI455X (gfx1250) — matrix operand order repaired and checked by reading, not compiled, not run on hardware
//
#include <hip/hip_runtime.h>
#include <stdint.h>

// E=8, T=16384, IN=2048, OUT=8192, CAP=2048 (fp32 grouped expert GEMM)
// fp32 emulated via split-bf16: x*y ~= hi*hi + hi*lo + lo*hi  (3 bf16 WMMAs / tile)

typedef __attribute__((ext_vector_type(16))) __bf16 v16bf;
typedef __attribute__((ext_vector_type(8)))  float  v8f;

#define NE    8
#define NIN   2048
#define NOUT  8192
#define CAP   2048
#define BM    128
#define BN    128
#define BK    32
#define KSTR  40                 // padded LDS row stride (bf16 elems)
#define PLANE (BM * KSTR)        // 5120 elems per plane
#define NITER (NIN / BK)         // 64 K-steps

// Pack two f32 into {hi16(x1),hi16(x0)} and residual pair with v_perm_b32.
__device__ __forceinline__ void split_pack(float x0, float x1, uint32_t& hi, uint32_t& lo) {
    uint32_t u0 = __float_as_uint(x0), u1 = __float_as_uint(x1);
    float h0 = __uint_as_float(u0 & 0xFFFF0000u);           // truncated bf16 (exact residual)
    float h1 = __uint_as_float(u1 & 0xFFFF0000u);
    uint32_t v0 = __float_as_uint(x0 - h0);
    uint32_t v1 = __float_as_uint(x1 - h1);
    hi = __builtin_amdgcn_perm(u1, u0, 0x07060302u);        // one v_perm per pair
    lo = __builtin_amdgcn_perm(v1, v0, 0x07060302u);
}

__device__ __forceinline__ void cvt_store(uint16_t* hiP, uint16_t* loP, int idx, float4 v) {
    uint32_t h01, l01, h23, l23;
    split_pack(v.x, v.y, h01, l01);
    split_pack(v.z, v.w, h23, l23);
    *reinterpret_cast<uint2*>(hiP + idx) = make_uint2(h01, h23);
    *reinterpret_cast<uint2*>(loP + idx) = make_uint2(l01, l23);
}

// A fragment (16x32 bf16): lane%16 = M row, lane/16 = K-half. Two b128 loads.
__device__ __forceinline__ v16bf load_a_frag(const uint16_t* p, int m, int sub) {
    union { uint4 q[2]; v16bf v; } u;
    u.q[0] = *reinterpret_cast<const uint4*>(p + m * KSTR + sub * 8);
    u.q[1] = *reinterpret_cast<const uint4*>(p + m * KSTR + 16 + sub * 8);
    return u.v;
}

// B fragment (32x16 bf16): lane%16 = N col, lane/16 = K-half. Two b128 loads.
__device__ __forceinline__ v16bf load_b_frag(const uint16_t* p, int n, int sub) {
    union { uint4 q[2]; v16bf v; } u;
    u.q[0] = *reinterpret_cast<const uint4*>(p + n * KSTR + sub * 8);
    u.q[1] = *reinterpret_cast<const uint4*>(p + n * KSTR + 16 + sub * 8);
    return u.v;
}

__global__ __launch_bounds__(256)
void experts_grouped_gemm(const float* __restrict__ in,
                          const float* __restrict__ wt,
                          const float* __restrict__ bias,
                          const int*   __restrict__ ef,
                          float*       __restrict__ out) {
    // double-buffered: 2 buffers x 4 planes (Ahi, Alo, Bhi, Blo) = 80 KB
    __shared__ __align__(16) uint16_t smem[2 * 4 * PLANE];

    const int tid = threadIdx.x;
    const int bid = blockIdx.x;
    const int e   = bid >> 10;
    const int mt  = (bid >> 6) & 15;
    const int nt  = bid & 63;

    int freq = 0, start = 0;
#pragma unroll
    for (int i = 0; i < NE; ++i) {
        int f = ef[i];
        if (i < e)  start += f;
        if (i == e) freq = f;
    }
    const int m0 = mt * BM;
    if (m0 >= freq) return;               // block-uniform early exit
    const int n0 = nt * BN;

    const int lane = tid & 31;
    const int w    = tid >> 5;
    const int wm   = w >> 1;
    const int wn   = w & 1;
    const int fm   = lane & 15;
    const int fsub = lane >> 4;

    v8f acc[2][4];
#pragma unroll
    for (int mi = 0; mi < 2; ++mi)
#pragma unroll
        for (int ni = 0; ni < 4; ++ni)
            acc[mi][ni] = (v8f){0.f, 0.f, 0.f, 0.f, 0.f, 0.f, 0.f, 0.f};

    // ---- loop-invariant staging geometry ----
    // Invalid A rows are merely CLAMPED (no zeroing needed): GEMM rows are
    // independent, so a garbage row only corrupts its own output row, which
    // the epilogue mask never stores. Clamping keeps loads in-bounds/finite.
    const float* aPtr[4];
    const float* bPtr[4];
    int idx[4];
#pragma unroll
    for (int i = 0; i < 4; ++i) {
        int f  = i * 256 + tid;
        int r  = f >> 3;                  // 0..127
        int c4 = (f & 7) * 4;             // 0..28
        idx[i] = r * KSTR + c4;
        int arow = m0 + r;
        int crow = (arow < freq) ? arow : (freq - 1);   // clamp -> always in-bounds
        aPtr[i] = in + (size_t)(start + crow) * NIN + c4;
        bPtr[i] = wt + ((size_t)e * NOUT + n0 + r) * NIN + c4;
    }

    // ---- prime the pipeline: prefetch tile 0 into registers ----
    float4 pa[4], pb[4];
#pragma unroll
    for (int i = 0; i < 4; ++i) {
        pa[i] = *reinterpret_cast<const float4*>(aPtr[i]);
        pb[i] = *reinterpret_cast<const float4*>(bPtr[i]);
    }

    for (int it = 0; it < NITER; ++it) {
        uint16_t* base = smem + (it & 1) * (4 * PLANE);
        uint16_t* sAhi = base;
        uint16_t* sAlo = base + PLANE;
        uint16_t* sBhi = base + 2 * PLANE;
        uint16_t* sBlo = base + 3 * PLANE;

        // convert current tile regs -> LDS (v_perm packing)
#pragma unroll
        for (int i = 0; i < 4; ++i) {
            cvt_store(sAhi, sAlo, idx[i], pa[i]);
            cvt_store(sBhi, sBlo, idx[i], pb[i]);
        }

        // prefetch next tile into registers (latency hidden behind WMMAs)
        if (it + 1 < NITER) {
            int ko = (it + 1) * BK;
#pragma unroll
            for (int i = 0; i < 4; ++i) {
                pa[i] = *reinterpret_cast<const float4*>(aPtr[i] + ko);
                pb[i] = *reinterpret_cast<const float4*>(bPtr[i] + ko);
            }
        }

        __syncthreads();                  // single barrier/iter (double-buffered)

        // ---- fragments ----
        v16bf ah[2], al[2], bh[4], bl[4];
#pragma unroll
        for (int mi = 0; mi < 2; ++mi) {
            int m = wm * 32 + mi * 16 + fm;
            ah[mi] = load_a_frag(sAhi, m, fsub);
            al[mi] = load_a_frag(sAlo, m, fsub);
        }
#pragma unroll
        for (int ni = 0; ni < 4; ++ni) {
            int n = wn * 64 + ni * 16 + fm;
            bh[ni] = load_b_frag(sBhi, n, fsub);
            bl[ni] = load_b_frag(sBlo, n, fsub);
        }

        // ---- split-bf16 WMMA (3 per 16x16 tile, 24 per wave per K-step) ----
#pragma unroll
        for (int mi = 0; mi < 2; ++mi) {
#pragma unroll
            for (int ni = 0; ni < 4; ++ni) {
                acc[mi][ni] = __builtin_amdgcn_wmma_f32_16x16x32_bf16(
                    false, ah[mi], false, bh[ni], (short)0, acc[mi][ni], false, false);
                acc[mi][ni] = __builtin_amdgcn_wmma_f32_16x16x32_bf16(
                    false, ah[mi], false, bl[ni], (short)0, acc[mi][ni], false, false);
                acc[mi][ni] = __builtin_amdgcn_wmma_f32_16x16x32_bf16(
                    false, al[mi], false, bh[ni], (short)0, acc[mi][ni], false, false);
            }
        }
    }

    // ---- epilogue: C/D layout lane%16 = N, lane/16 selects M+8, VGPR r = M row ----
    const bool full = (m0 + BM) <= freq;  // block-uniform: all rows valid
    if (full) {
#pragma unroll
        for (int ni = 0; ni < 4; ++ni) {
            int col = n0 + wn * 64 + ni * 16 + fm;
            float bv = bias[(size_t)e * NOUT + col];
#pragma unroll
            for (int mi = 0; mi < 2; ++mi) {
                v8f a = acc[mi][ni];
#pragma unroll
                for (int r = 0; r < 8; ++r) {
                    int mrow = m0 + wm * 32 + mi * 16 + fsub * 8 + r;
                    out[(size_t)(start + mrow) * NOUT + col] = a[r] + bv;
                }
            }
        }
    } else {
#pragma unroll
        for (int ni = 0; ni < 4; ++ni) {
            int col = n0 + wn * 64 + ni * 16 + fm;
            float bv = bias[(size_t)e * NOUT + col];
#pragma unroll
            for (int mi = 0; mi < 2; ++mi) {
                v8f a = acc[mi][ni];
#pragma unroll
                for (int r = 0; r < 8; ++r) {
                    int mrow = m0 + wm * 32 + mi * 16 + fsub * 8 + r;
                    if (mrow < freq)
                        out[(size_t)(start + mrow) * NOUT + col] = a[r] + bv;
                }
            }
        }
    }
}

extern "C" void kernel_launch(void* const* d_in, const int* in_sizes, int n_in,
                              void* d_out, int out_size, void* d_ws, size_t ws_size,
                              hipStream_t stream) {
    const float* in   = (const float*)d_in[0];
    const float* wt   = (const float*)d_in[1];
    const float* bs   = (const float*)d_in[2];
    const int*   ef   = (const int*)d_in[3];
    float*       outp = (float*)d_out;

    dim3 grid(NE * (CAP / BM) * (NOUT / BN));   // 8192 blocks, 256 threads (8 wave32)
    experts_grouped_gemm<<<grid, 256, 0, stream>>>(in, wt, bs, ef, outp);
}
